// HANLayer_17592186044981
// MI455X (gfx1250) — compile-verified, matrix operand order checked by reading, not run on hardware
//
#include <hip/hip_runtime.h>
#include <math.h>

typedef __attribute__((ext_vector_type(16))) _Float16 v16h;
typedef __attribute__((ext_vector_type(8)))  float    v8f;

#define NNODES 100000
#define NEDGES 500000
#define FDIM   128
#define NHEAD  8

// ---------------------------------------------------------------------------
// WMMA 16x16x32 f16 fragment-order index for a 128x128 weight matrix.
// Fragment storage: [ntile(8)][kstep(4)][lane(32)][e(16)] halves, so each
// lane's 16-half fragment is contiguous (one v16h = 2x b128 loads).
// A/B 16-bit layout (ISA 7.12.2): lane = (n&15) + 16*hi, hi = (k%16)>=8,
// e = 8*(k%32>=16) + (k%8).
// ---------------------------------------------------------------------------
__device__ __forceinline__ int frag_pack_idx(int k, int n)
{
    const int ks    = k >> 5;
    const int kr    = k & 31;
    const int hi    = (kr >> 3) & 1;
    const int e     = ((kr >> 4) << 3) | (kr & 7);
    const int ntile = n >> 4;
    const int lane  = (n & 15) | (hi << 4);
    return ((((ntile << 2) + ks) << 5) + lane) * 16 + e;
}

// ---------------------------------------------------------------------------
// Fold per-head relation transforms into K/V projection weights (f16, packed):
//   Wc[r][k][h*16+e] = sum_d W[st][k][h*16+d] * rel[r][h][d][e]
// grid: 3*128 blocks (r,k), 128 threads (n)
// ---------------------------------------------------------------------------
__global__ __launch_bounds__(128) void fold_weights(
    const float* __restrict__ Wk, const float* __restrict__ bk,
    const float* __restrict__ Wv, const float* __restrict__ bv,
    const float* __restrict__ rel_att, const float* __restrict__ rel_msg,
    _Float16* __restrict__ Wkc, _Float16* __restrict__ Wvc,
    float* __restrict__ bkc, float* __restrict__ bvc)
{
    const int r = blockIdx.x >> 7;
    const int k = blockIdx.x & 127;
    const int n = threadIdx.x;
    const int h = n >> 4, e = n & 15;
    const int st = (r == 1) ? 1 : 0;   // rel 0,2: src type a; rel 1: type b
    const float* wkrow = Wk + (size_t)st*FDIM*FDIM + (size_t)k*FDIM + h*16;
    const float* wvrow = Wv + (size_t)st*FDIM*FDIM + (size_t)k*FDIM + h*16;
    const float* att   = rel_att + (size_t)((r*NHEAD + h)*16)*16;   // [d][e]
    const float* msg   = rel_msg + (size_t)((r*NHEAD + h)*16)*16;
    float accK = 0.f, accV = 0.f;
#pragma unroll
    for (int d = 0; d < 16; ++d) {
        accK += wkrow[d] * att[d*16 + e];
        accV += wvrow[d] * msg[d*16 + e];
    }
    const int p = frag_pack_idx(k, n);
    Wkc[(size_t)r*FDIM*FDIM + p] = (_Float16)accK;
    Wvc[(size_t)r*FDIM*FDIM + p] = (_Float16)accV;
    if (k == 0) {
        float bK = 0.f, bV = 0.f;
#pragma unroll
        for (int d = 0; d < 16; ++d) {
            bK += bk[st*FDIM + h*16 + d] * att[d*16 + e];
            bV += bv[st*FDIM + h*16 + d] * msg[d*16 + e];
        }
        bkc[r*FDIM + n] = bK;
        bvc[r*FDIM + n] = bV;
    }
}

// pack plain f32 [nmats][128][128] weights into packed f16 fragment order
__global__ __launch_bounds__(256) void pack_half(const float* __restrict__ src,
                                                 _Float16* __restrict__ dst, int nmats)
{
    const int i = blockIdx.x*blockDim.x + threadIdx.x;
    if (i >= nmats * FDIM * FDIM) return;
    const int mat = i >> 14, kn = i & 16383;
    const int k = kn >> 7, n = kn & 127;
    dst[(size_t)mat*FDIM*FDIM + frag_pack_idx(k, n)] = (_Float16)src[i];
}

__global__ __launch_bounds__(256) void fill_f(float* __restrict__ p, float v, size_t n)
{
    for (size_t i = blockIdx.x*(size_t)blockDim.x + threadIdx.x; i < n;
         i += (size_t)gridDim.x*blockDim.x)
        p[i] = v;
}

// invert softmax denominators once (saves a div chain per (edge,head))
__global__ __launch_bounds__(256) void recip_f(float* __restrict__ p, size_t n)
{
    for (size_t i = blockIdx.x*(size_t)blockDim.x + threadIdx.x; i < n;
         i += (size_t)gridDim.x*blockDim.x) {
        const float v = p[i];
        p[i] = (v > 0.0f) ? __fdividef(1.0f, v) : 0.0f;
    }
}

// ---------------------------------------------------------------------------
// WMMA GEMM: out[rows x 128] = A[rows x 128] (f32, scaled) @ W(packed f16) + bias
// optional epilogue: out = out*sigmoid(skip[idx]) + blend*(1-sigmoid(skip[idx]))
// 256 threads = 8 wave32; block owns a 32-row strip (2 A tiles); wave w owns
// cols [16w,16w+16). Per k-step: each B fragment is loaded once and feeds two
// WMMAs (A0*B -> c0, A1*B -> c1), halving weight traffic.
// ---------------------------------------------------------------------------
__global__ __launch_bounds__(256) void gemm_n128_wmma(
    const float* __restrict__ A, const _Float16* __restrict__ W,
    const float* __restrict__ bias, float* __restrict__ out, int rows,
    const float* __restrict__ blend, const float* __restrict__ skip, int skipIdx,
    float inScale)
{
    __shared__ _Float16 ldsA[2 * 2048];   // two 16x128 tiles, fragment order (8KB)
    const int tid  = threadIdx.x;
    const int row0 = blockIdx.x << 5;

    __builtin_prefetch(W + (size_t)(tid >> 5) * 512, 0, 3);          // weight frags
    if (blend) __builtin_prefetch(blend + (size_t)row0 * FDIM, 0, 1);

    // cooperative f32 -> f16 conversion, stored directly in fragment order
    for (int i = tid; i < 32 * FDIM; i += 256) {
        const int m = i >> 7, k = i & 127;          // m in 0..31
        const int row = row0 + m;
        const float v = (row < rows) ? A[(size_t)row*FDIM + k] * inScale : 0.0f;
        const int ks = k >> 5, kr = k & 31;
        const int hi = (kr >> 3) & 1;
        const int e  = ((kr >> 4) << 3) | (kr & 7);
        const int tile = m >> 4, mm = m & 15;
        ldsA[tile*2048 + (((((ks << 5) + (mm | (hi << 4)))) << 4) | e)] = (_Float16)v;
    }
    __syncthreads();

    const int wave = tid >> 5;
    const int lane = tid & 31;
    const v16h* Ap = (const v16h*)ldsA;                 // tile0: [ks*32+lane], tile1: +128
    const v16h* Bp = ((const v16h*)W) + (wave << 7);    // wave's 4*32 fragments

    v8f c0 = {0.f,0.f,0.f,0.f,0.f,0.f,0.f,0.f};
    v8f c1 = {0.f,0.f,0.f,0.f,0.f,0.f,0.f,0.f};
#pragma unroll
    for (int ks = 0; ks < 4; ++ks) {
        const v16h b  = Bp[(ks << 5) + lane];
        const v16h a0 = Ap[(ks << 5) + lane];
        const v16h a1 = Ap[128 + (ks << 5) + lane];
        c0 = __builtin_amdgcn_wmma_f32_16x16x32_f16(false, a0, false, b, (short)0, c0,
                                                    false, false);
        c1 = __builtin_amdgcn_wmma_f32_16x16x32_f16(false, a1, false, b, (short)0, c1,
                                                    false, false);
    }

    const int n = (wave << 4) + (lane & 15);
    float alpha = 1.0f;
    if (skip) {
        const float s = skip[skipIdx];
        alpha = __fdividef(1.0f, 1.0f + __expf(-s));
    }
    const float bn    = bias ? bias[n] : 0.0f;
    const int   mbase = (lane >> 4) << 3;   // C layout: VGPR j -> M=j (+8 hi half)
    v8f cc[2] = {c0, c1};
#pragma unroll
    for (int tile = 0; tile < 2; ++tile) {
#pragma unroll
        for (int j = 0; j < 8; ++j) {
            const int row = row0 + tile*16 + mbase + j;
            if (row < rows) {
                float v = cc[tile][j] + bn;
                if (blend) v = v*alpha + blend[(size_t)row*FDIM + n]*(1.0f - alpha);
                out[(size_t)row*FDIM + n] = v;
            }
        }
    }
}

// ---------------------------------------------------------------------------
// Edge-softmax passes (one thread per (edge, head); tables are L2-resident)
// ---------------------------------------------------------------------------
__device__ inline void atomicMaxF(float* addr, float val)
{
    if (val >= 0.0f) atomicMax((int*)addr, __float_as_int(val));
    else             atomicMin((unsigned int*)addr, (unsigned int)__float_as_int(val));
}

__global__ __launch_bounds__(256) void edge_score(
    const float* __restrict__ Q, const float* __restrict__ K,
    const int* __restrict__ src, const int* __restrict__ dst,
    const float* __restrict__ pri, float* __restrict__ score,
    float* __restrict__ smax, int E)
{
    const int idx = blockIdx.x*blockDim.x + threadIdx.x;
    const int e = idx >> 3, h = idx & 7;
    if (e >= E) return;
    const int s = src[e], d = dst[e];
    const float4* qp = (const float4*)(Q + (size_t)d*FDIM + h*16);
    const float4* kp = (const float4*)(K + (size_t)s*FDIM + h*16);
    float acc = 0.f;
#pragma unroll
    for (int i = 0; i < 4; ++i) {
        const float4 q4 = qp[i], k4 = kp[i];
        acc += q4.x*k4.x + q4.y*k4.y + q4.z*k4.z + q4.w*k4.w;
    }
    const float sc = acc * pri[h] * 0.25f;     // / sqrt(DK=16)
    score[idx] = sc;
    atomicMaxF(&smax[(size_t)d*NHEAD + h], sc);
}

__global__ __launch_bounds__(256) void edge_exp(
    float* __restrict__ score, const float* __restrict__ smax,
    const int* __restrict__ dst, float* __restrict__ den, int E)
{
    const int idx = blockIdx.x*blockDim.x + threadIdx.x;
    const int e = idx >> 3, h = idx & 7;
    if (e >= E) return;
    const int d = dst[e];
    const float ex = __expf(score[idx] - smax[(size_t)d*NHEAD + h]);
    score[idx] = ex;
    atomicAdd(&den[(size_t)d*NHEAD + h], ex);
}

// rden holds 1/den after recip_f
__global__ __launch_bounds__(256) void edge_agg(
    const float* __restrict__ ex, const float* __restrict__ rden,
    const float* __restrict__ V, const int* __restrict__ src,
    const int* __restrict__ dst, float* __restrict__ t, int E)
{
    const int idx = blockIdx.x*blockDim.x + threadIdx.x;
    const int e = idx >> 3, h = idx & 7;
    if (e >= E) return;
    const int s = src[e], d = dst[e];
    const float a = ex[idx] * rden[(size_t)d*NHEAD + h];
    const float4* vp = (const float4*)(V + (size_t)s*FDIM + h*16);
    float* tp = t + (size_t)d*FDIM + h*16;
#pragma unroll
    for (int i = 0; i < 4; ++i) {
        const float4 v4 = vp[i];
        atomicAdd(tp + i*4 + 0, v4.x*a);
        atomicAdd(tp + i*4 + 1, v4.y*a);
        atomicAdd(tp + i*4 + 2, v4.z*a);
        atomicAdd(tp + i*4 + 3, v4.w*a);
    }
}

// ---------------------------------------------------------------------------
extern "C" void kernel_launch(void* const* d_in, const int* in_sizes, int n_in,
                              void* d_out, int out_size, void* d_ws, size_t ws_size,
                              hipStream_t stream)
{
    (void)in_sizes; (void)n_in; (void)out_size;
    const int N = NNODES, E = NEDGES;
    const size_t NF = (size_t)N * FDIM;
    const size_t EH = (size_t)E * NHEAD;
    const size_t NH = (size_t)N * NHEAD;

    const float* h_a = (const float*)d_in[0];
    const float* h_b = (const float*)d_in[1];
    const float* Wk  = (const float*)d_in[8];
    const float* bk  = (const float*)d_in[9];
    const float* Wv  = (const float*)d_in[10];
    const float* bv  = (const float*)d_in[11];
    const float* Wq  = (const float*)d_in[12];
    const float* bq  = (const float*)d_in[13];
    const float* Wa  = (const float*)d_in[14];
    const float* ba  = (const float*)d_in[15];
    const float* rel_att = (const float*)d_in[16];
    const float* rel_msg = (const float*)d_in[17];
    const float* rel_pri = (const float*)d_in[18];
    const float* skip    = (const float*)d_in[19];
    float* out = (float*)d_out;

    // workspace layout (floats, then packed f16 weights)
    float* ws    = (float*)d_ws;
    float* Qa    = ws;
    float* Qb    = Qa + NF;
    float* Kr    = Qb + NF;
    float* Vr    = Kr + NF;
    float* ta    = Vr + NF;
    float* tb    = ta + NF;
    float* score = tb + NF;           // EH
    float* smax  = score + EH;        // NH
    float* den   = smax + NH;         // NH
    float* bkc   = den + NH;          // 3*128
    float* bvc   = bkc + 3*FDIM;      // 3*128
    _Float16* Wh = (_Float16*)(bvc + 3*FDIM);
    // Wh blocks of 16384: [0..2]=Wkc[r], [3..5]=Wvc[r], [6..7]=Wq[t], [8..9]=Wa[t]
    const size_t WSZ = (size_t)FDIM * FDIM;
    const size_t required = (size_t)((char*)(Wh + 10*WSZ) - (char*)d_ws);
    if (ws_size < required) return;

    // 1) fold relation transforms into packed f16 GEMM weights
    fold_weights<<<dim3(3*FDIM), dim3(FDIM), 0, stream>>>(
        Wk, bk, Wv, bv, rel_att, rel_msg, Wh, Wh + 3*WSZ, bkc, bvc);
    pack_half<<<128, 256, 0, stream>>>(Wq, Wh + 6*WSZ, 2);
    pack_half<<<128, 256, 0, stream>>>(Wa, Wh + 8*WSZ, 2);

    const int GB = (N + 31) / 32;     // 3125 GEMM blocks (32 rows each)
    // 2) Q projections (dst-type): Qa = h_a@Wq[0]+bq[0], Qb = h_b@Wq[1]+bq[1]
    gemm_n128_wmma<<<GB, 256, 0, stream>>>(h_a, Wh + 6*WSZ, bq,        Qa, N,
                                           nullptr, nullptr, 0, 1.0f);
    gemm_n128_wmma<<<GB, 256, 0, stream>>>(h_b, Wh + 7*WSZ, bq + FDIM, Qb, N,
                                           nullptr, nullptr, 0, 1.0f);

    fill_f<<<2048, 256, 0, stream>>>(ta, 0.0f, NF);
    fill_f<<<2048, 256, 0, stream>>>(tb, 0.0f, NF);

    const int EB = (int)((EH + 255) / 256);   // 15625
    for (int r = 0; r < 3; ++r) {
        const float* xsrc = (r == 1) ? h_b : h_a;
        const int*   srcI = (const int*)d_in[2 + 2*r];
        const int*   dstI = (const int*)d_in[3 + 2*r];
        const float* Qd   = (r == 1) ? Qa : Qb;
        float*       tt   = (r == 1) ? ta : tb;

        // 3) folded K/V projections for this relation
        gemm_n128_wmma<<<GB, 256, 0, stream>>>(xsrc, Wh + (size_t)r*WSZ,
                                               bkc + r*FDIM, Kr, N,
                                               nullptr, nullptr, 0, 1.0f);
        gemm_n128_wmma<<<GB, 256, 0, stream>>>(xsrc, Wh + (size_t)(3+r)*WSZ,
                                               bvc + r*FDIM, Vr, N,
                                               nullptr, nullptr, 0, 1.0f);

        // 4) segment softmax + weighted scatter
        fill_f<<<1024, 256, 0, stream>>>(smax, -INFINITY, NH);
        fill_f<<<1024, 256, 0, stream>>>(den, 0.0f, NH);
        edge_score<<<EB, 256, 0, stream>>>(Qd, Kr, srcI, dstI,
                                           rel_pri + r*NHEAD, score, smax, E);
        edge_exp<<<EB, 256, 0, stream>>>(score, smax, dstI, den, E);
        recip_f<<<1024, 256, 0, stream>>>(den, NH);
        edge_agg<<<EB, 256, 0, stream>>>(score, den, Vr, srcI, dstI, tt, E);
    }

    // 5) output projection + sigmoid-skip blend; t_b gets the 0.5 mean factor
    gemm_n128_wmma<<<GB, 256, 0, stream>>>(ta, Wh + 8*WSZ, ba,        out,      N,
                                           h_a, skip, 0, 1.0f);
    gemm_n128_wmma<<<GB, 256, 0, stream>>>(tb, Wh + 9*WSZ, ba + FDIM, out + NF, N,
                                           h_b, skip, 1, 0.5f);
}
